// entropy_parameters_38972533243998
// MI455X (gfx1250) — matrix operand order repaired and checked by reading, not compiled, not run on hardware
//
#include <hip/hip_runtime.h>

// ---------------------------------------------------------------------------
// 3x quantized 1x1-conv (GEMM) chain for MI455X (gfx1250, wave32, WMMA).
//
// sum_{i=0}^{3} floor((n+i)/4) == n  (integer n)  =>  the reference's split
// conv is exactly  out = x @ round(w)^T + round(b).
// x in [-128,127] and round(w) (|w| <~ 200) are exact in fp16, so
// V_WMMA_F32_16X16X32_F16 with f32 accumulation matches the reference's fp32
// einsum arithmetic. Epilogue is exact int64 fixed-point.
//
// GEMM: block tile 128x128 (256 thr / 8 waves), wave tile 64x32,
// double-buffered LDS staged with GLOBAL_LOAD_ASYNC_TO_LDS_B128 (ASYNCcnt).
// ---------------------------------------------------------------------------

typedef _Float16 v8h  __attribute__((ext_vector_type(8)));
typedef _Float16 v16h __attribute__((ext_vector_type(16)));
typedef float    v8f  __attribute__((ext_vector_type(8)));
typedef int      v4i  __attribute__((vector_size(16)));   // matches builtin param

#define NB    8
#define HW    4096
#define MDIM  32768   // NB * HW

#define BM 128
#define BN 128
#define BK 32
#define LDT 40        // padded LDS row stride in halves (80B, 16B aligned)

#if defined(__AMDGCN__) && __has_builtin(__builtin_amdgcn_global_load_async_to_lds_b128)
#define USE_ASYNC_COPY 1
#else
#define USE_ASYNC_COPY 0
#endif

// 16-byte global -> LDS copy (async on CDNA5, generic fallback otherwise)
__device__ __forceinline__ void async_cp16(const _Float16* g, _Float16* l) {
#if USE_ASYNC_COPY
    typedef __attribute__((address_space(1))) v4i* gp_t;   // global, printed __device__
    typedef __attribute__((address_space(3))) v4i* lp_t;   // LDS (32-bit offset ptr)
    // generic->AS1: same 64-bit representation; generic->AS3: low 32 bits of a
    // generic LDS pointer are the LDS byte offset (ISA 10.2 aperture rule).
    __builtin_amdgcn_global_load_async_to_lds_b128(
        (gp_t)(unsigned long long)(size_t)g,
        (lp_t)(unsigned)(size_t)l,
        /*offset=*/0, /*cpol=*/0);
#else
    *(v8h*)l = *(const v8h*)g;
#endif
}

template <int N>
__device__ __forceinline__ void wait_async() {
#if USE_ASYNC_COPY
# if __has_builtin(__builtin_amdgcn_s_wait_asynccnt)
    __builtin_amdgcn_s_wait_asynccnt(N);
# else
    asm volatile("s_wait_asynccnt %0" :: "i"(N) : "memory");
# endif
#endif
}

// ---------------------------------------------------------------------------
// round(w) -> fp16, elementwise
// ---------------------------------------------------------------------------
__global__ void wquant_kernel(const float* __restrict__ w,
                              _Float16* __restrict__ wq, int n) {
    int i = blockIdx.x * blockDim.x + threadIdx.x;
    if (i < n) wq[i] = (_Float16)rintf(w[i]);
}

// ---------------------------------------------------------------------------
// x [8][768][4096] f32 NCHW  ->  A [32768][768] fp16 row-major (M-major),
// with clip(-128,127). LDS 32x32 transpose so both global sides coalesce.
// ---------------------------------------------------------------------------
__global__ __launch_bounds__(256)
void pack_x_kernel(const float* __restrict__ x, _Float16* __restrict__ A) {
    __shared__ float tile[32][33];
    const int b   = blockIdx.z;
    const int hw0 = blockIdx.x * 32;
    const int c0  = blockIdx.y * 32;
    const int tx  = threadIdx.x;       // 0..31
    const int ty  = threadIdx.y;       // 0..7

    const float* xp = x + ((size_t)b * 768 + c0) * HW + hw0;
    #pragma unroll
    for (int j = 0; j < 32; j += 8)
        tile[ty + j][tx] = xp[(size_t)(ty + j) * HW + tx];   // coalesced in hw
    __syncthreads();

    _Float16* Ap = A + ((size_t)b * HW + hw0) * 768 + c0;
    #pragma unroll
    for (int j = 0; j < 32; j += 8) {
        float v = tile[tx][ty + j];
        v = fminf(fmaxf(v, -128.0f), 127.0f);
        Ap[(size_t)(ty + j) * 768 + tx] = (_Float16)v;        // coalesced in c
    }
}

// ---------------------------------------------------------------------------
// GEMM: D[m][n] = sum_k A[m][k] * Wq[n][k], then quantization epilogue.
//   A  : [MDIM][K] fp16 row-major
//   Wq : [N][K]    fp16 row-major (natural w[o][c] layout)
// 256 thr / 8 waves; block tile 128x128; wave tile 64x32 (4x2 subtiles).
// Double-buffered LDS, async global->LDS staging.
// mode 0: t=(a*mul+rnd)>>shift; clip[0,clp]; t=(t*scl+2^20)>>21; -> fp16 [M][N]
// mode 1: t=(a*mul+rnd)>>shift; -> f32 NCHW [8][N][4096]
// ---------------------------------------------------------------------------
__global__ __launch_bounds__(256)
void qgemm_kernel(const _Float16* __restrict__ A,
                  const _Float16* __restrict__ Wq,
                  const float* __restrict__ bias,
                  const float* __restrict__ muls,
                  const long long* __restrict__ dvds, int dvd_idx, int shift_adj,
                  const float* __restrict__ relus, int relu_idx,
                  int K, int N, int mode,
                  _Float16* __restrict__ outH, float* __restrict__ outF)
{
    __shared__ __align__(16) _Float16 As[2 * BM * LDT];
    __shared__ __align__(16) _Float16 Bs[2 * BN * LDT];

    const int tid   = threadIdx.x;
    const int lane  = tid & 31;
    const int wid   = tid >> 5;           // 0..7
    const int m0    = blockIdx.x * BM;
    const int n0    = blockIdx.y * BN;
    const int waveM = (wid & 1) * 64;     // 0 or 64
    const int waveN = (wid >> 1) * 32;    // 0,32,64,96
    const int arow  = lane & 15;          // row within 16-tile (ISA 7.12.2)
    const int khalf = lane >> 4;          // which K-half this lane holds

    v8f c[4][2];
    #pragma unroll
    for (int i = 0; i < 4; ++i)
        #pragma unroll
        for (int j = 0; j < 2; ++j)
            #pragma unroll
            for (int e = 0; e < 8; ++e) c[i][j][e] = 0.0f;

    const int ldr = tid >> 2;             // 0..63: row within a load pass
    const int ldk = (tid & 3) * 8;        // 0,8,16,24 halves (16B chunks)

    const int nsteps = K / BK;

    // ---- prologue: stage k-tile 0 into buffer 0 (4 async b128 per thread)
    {
        #pragma unroll
        for (int p = 0; p < 2; ++p) {
            int r = p * 64 + ldr;
            async_cp16(A  + (size_t)(m0 + r) * K + ldk, As + r * LDT + ldk);
            async_cp16(Wq + (size_t)(n0 + r) * K + ldk, Bs + r * LDT + ldk);
        }
    }

    for (int s = 0; s < nsteps; ++s) {
        const int cur  = s & 1;
        const int nxt  = cur ^ 1;
        const int kk   = (s + 1) * BK;
        const bool hasNext = (s + 1 < nsteps);

        if (hasNext) {
            // prefetch next k-tile into the other buffer (safe: last read of
            // that buffer was protected by the previous end-of-iter barrier)
            #pragma unroll
            for (int p = 0; p < 2; ++p) {
                int r = p * 64 + ldr;
                async_cp16(A  + (size_t)(m0 + r) * K + kk + ldk,
                           As + nxt * (BM * LDT) + r * LDT + ldk);
                async_cp16(Wq + (size_t)(n0 + r) * K + kk + ldk,
                           Bs + nxt * (BN * LDT) + r * LDT + ldk);
            }
            wait_async<4>();   // retire current tile's copies (in-order)
        } else {
            wait_async<0>();
        }
        __syncthreads();

        const _Float16* Asb = As + cur * (BM * LDT);
        const _Float16* Bsb = Bs + cur * (BN * LDT);

        // B fragments (32x16): lane = col (lane&15), lanes<16 hold K[0:8]+K[16:24],
        // lanes>=16 hold K[8:16]+K[24:32] (same K order as A).
        v16h bfrag[2];
        #pragma unroll
        for (int j = 0; j < 2; ++j) {
            const _Float16* bp = Bsb + (waveN + j * 16 + arow) * LDT + khalf * 8;
            v8h lo = *(const v8h*)bp;
            v8h hi = *(const v8h*)(bp + 16);
            #pragma unroll
            for (int e = 0; e < 8; ++e) { bfrag[j][e] = lo[e]; bfrag[j][8 + e] = hi[e]; }
        }

        // A fragments (16x32): lanes<16 hold K[0:8]+K[16:24],
        // lanes>=16 hold K[8:16]+K[24:32]  (ISA 7.12.2 16-bit A layout)
        #pragma unroll
        for (int i = 0; i < 4; ++i) {
            const _Float16* ap = Asb + (waveM + i * 16 + arow) * LDT + khalf * 8;
            v8h lo = *(const v8h*)ap;
            v8h hi = *(const v8h*)(ap + 16);
            v16h afrag;
            #pragma unroll
            for (int e = 0; e < 8; ++e) { afrag[e] = lo[e]; afrag[8 + e] = hi[e]; }
            #pragma unroll
            for (int j = 0; j < 2; ++j)
                c[i][j] = __builtin_amdgcn_wmma_f32_16x16x32_f16(
                    /*neg_a=*/false, afrag, /*neg_b=*/false, bfrag[j],
                    /*c_mod=*/(short)0, c[i][j],
                    /*reuse_a=*/false, /*reuse_b=*/false);
        }
        __syncthreads();  // all waves done reading `cur` before it is restaged
    }

    // ---------------- epilogue (exact int64 fixed point) ----------------
    const int shift = (int)dvds[dvd_idx] + shift_adj;
    const long long rnd = 1LL << (shift - 1);
    long long clpI = 0, sclI = 0;
    if (mode == 0) {
        float relu = relus[relu_idx];
        clpI = (long long)llrintf(255.0f / relu * 33554432.0f); // 2^(16+9)
        sclI = (long long)floorf((relu + 8.0f) * 0.0625f);      // floor((relu+8)/16)
    }

    #pragma unroll
    for (int j = 0; j < 2; ++j) {
        const int n = n0 + waveN + j * 16 + arow;
        const long long bI = (long long)llrintf(bias[n]);
        const long long mI = (long long)llrintf(muls[n]);
        #pragma unroll
        for (int i = 0; i < 4; ++i) {
            const int mb = m0 + waveM + i * 16 + khalf * 8;  // C/D: M = r + 8*(lane>=16)
            #pragma unroll
            for (int r = 0; r < 8; ++r) {
                const int m = mb + r;
                long long a = (long long)llrintf(c[i][j][r]) + bI;
                long long t = (a * mI + rnd) >> shift;       // floor division
                if (mode == 0) {
                    t = t < 0 ? 0 : (t > clpI ? clpI : t);
                    t = (t * sclI + (1LL << 20)) >> 21;
                    outH[(size_t)m * N + n] = (_Float16)(float)t;
                } else {
                    const int b  = m >> 12;       // /4096
                    const int hw = m & 4095;
                    outF[((size_t)b * N + n) * HW + hw] = (float)t;
                }
            }
        }
    }
}

// ---------------------------------------------------------------------------
extern "C" void kernel_launch(void* const* d_in, const int* in_sizes, int n_in,
                              void* d_out, int out_size, void* d_ws, size_t ws_size,
                              hipStream_t stream) {
    const float*     x     = (const float*)d_in[0];
    const float*     w1    = (const float*)d_in[1];
    const float*     b1    = (const float*)d_in[2];
    const float*     w2    = (const float*)d_in[3];
    const float*     b2    = (const float*)d_in[4];
    const float*     w3    = (const float*)d_in[5];
    const float*     b3    = (const float*)d_in[6];
    const float*     muls0 = (const float*)d_in[7];
    const float*     muls1 = (const float*)d_in[8];
    const float*     muls2 = (const float*)d_in[9];
    const long long* dvds  = (const long long*)d_in[10];
    const float*     relus = (const float*)d_in[11];

    // workspace layout (bytes):
    //   [0,            50331648)  A1 (32768x768 fp16)  -- later reused as A3
    //   [50331648,     92274688)  A2 (32768x640 fp16)
    //   [92274688,     93257728)  W1q (640x768 fp16)
    //   [93257728,     93913088)  W2q (512x640 fp16)
    //   [93913088,     94306304)  W3q (384x512 fp16)
    char* ws = (char*)d_ws;
    _Float16* A1  = (_Float16*)(ws);
    _Float16* A2  = (_Float16*)(ws + 50331648u);
    _Float16* A3  = (_Float16*)(ws);              // A1 dead after layer 1
    _Float16* W1q = (_Float16*)(ws + 92274688u);
    _Float16* W2q = (_Float16*)(ws + 93257728u);
    _Float16* W3q = (_Float16*)(ws + 93913088u);

    // 1) quantize weights
    {
        int n1 = 640 * 768, n2 = 512 * 640, n3 = 384 * 512;
        wquant_kernel<<<(n1 + 255) / 256, 256, 0, stream>>>(w1, W1q, n1);
        wquant_kernel<<<(n2 + 255) / 256, 256, 0, stream>>>(w2, W2q, n2);
        wquant_kernel<<<(n3 + 255) / 256, 256, 0, stream>>>(w3, W3q, n3);
    }

    // 2) pack/transpose activations NCHW f32 -> [M][768] fp16
    {
        dim3 grid(HW / 32, 768 / 32, NB);
        dim3 block(32, 8);
        pack_x_kernel<<<grid, block, 0, stream>>>(x, A1);
    }

    // 3) layer 1: 32768 x 640 x 768, shift = d0 + HS_LEFT - CLP_K = d0 - 7
    qgemm_kernel<<<dim3(MDIM / BM, 640 / BN), 256, 0, stream>>>(
        A1, W1q, b1, muls0, dvds, 0, -7, relus, 0,
        768, 640, 0, A2, nullptr);

    // 4) layer 2: 32768 x 512 x 640, shift = d1 - CLP_K = d1 - 9
    qgemm_kernel<<<dim3(MDIM / BM, 512 / BN), 256, 0, stream>>>(
        A2, W2q, b2, muls1, dvds, 1, -9, relus, 1,
        640, 512, 0, A3, nullptr);

    // 5) layer 3: 32768 x 384 x 512, shift = d2 - EN_LEFT = d2 - 4 -> f32 NCHW
    qgemm_kernel<<<dim3(MDIM / BM, 384 / BN), 256, 0, stream>>>(
        A3, W3q, b3, muls2, dvds, 2, -4, relus, 0,
        512, 384, 1, nullptr, (float*)d_out);
}
